// SpeakingEncoder_87754771792645
// MI455X (gfx1250) — matrix operand order repaired and checked by reading, not compiled, not run on hardware
//
#include <hip/hip_runtime.h>

// ---------------------------------------------------------------------------
// VQ encoder for MI455X (gfx1250, wave32, WMMA + Tensor Data Mover).
//
// score(row, n) = ||c_n||^2 - 2 * <x_row, c_n>   (x^2 term dropped: argmin-inv)
// <x,c> via bf16x3 split-precision WMMA (xh.ch + xh.cl + xl.ch, f32 accum).
// Fused argmin epilogue avoids the 512 MB score tensor.
// LDS staging via TDM (tensor_load_to_lds) with double buffering:
//   wave0 issues 4 TDM tile loads for chunk k+1 while all waves WMMA chunk k,
//   then s_wait_tensorcnt 0 + one barrier. D# pad fields add the 16B/row LDS
//   padding in hardware (32-elem rows: pad_interval=16DW, pad_amount=4DW).
// ---------------------------------------------------------------------------

typedef __attribute__((ext_vector_type(16))) __bf16 v16bf;
typedef __attribute__((ext_vector_type(8)))  float  v8f;
typedef __attribute__((ext_vector_type(4)))  unsigned int u32x4;
typedef __attribute__((ext_vector_type(8)))  int    i32x8;
typedef __attribute__((ext_vector_type(4)))  int    i32x4;

#define SEQ      2048
#define BATCH    8
#define DIM      512
#define NCODES   8192
#define M_TOTAL  (SEQ * BATCH)     // 16384 rows

#define BM 64        // rows per workgroup
#define BN 128       // cols per N tile
#define BK 32        // K chunk (one bf16 WMMA K-step)
#define A_LD 40      // 32 + 8 pad (ushort units) -> 80 B/row, TDM pad fields
#define B_LD 40

#define ASIZE_PLANE (BM * A_LD * 2)   //  5120 B per (buf,ver) A plane
#define BSIZE_PLANE (BN * B_LD * 2)   // 10240 B per (buf,ver) B plane
#define SMEM_BYTES  (4 * ASIZE_PLANE + 4 * BSIZE_PLANE)   // 61440 B

union Frag { uint4 q[2]; v16bf v; };

// --------------------------- prep kernels ----------------------------------

__global__ __launch_bounds__(256) void split_x_kernel(
    const float* __restrict__ x, __bf16* __restrict__ Ah,
    __bf16* __restrict__ Al, int n) {
  int i = blockIdx.x * blockDim.x + threadIdx.x;
  int stride = gridDim.x * blockDim.x;
  for (; i < n; i += stride) {
    float v = x[i];
    __bf16 h = (__bf16)v;
    Ah[i] = h;
    Al[i] = (__bf16)(v - (float)h);
  }
}

__global__ __launch_bounds__(256) void split_codebook_kernel(
    const float* __restrict__ cb, __bf16* __restrict__ Bh,
    __bf16* __restrict__ Bl, float* __restrict__ csq) {
  __shared__ float red[256];
  const int n = blockIdx.x;
  const int t = threadIdx.x;
  float ssum = 0.f;
  for (int d = t; d < DIM; d += 256) {
    float v = cb[(size_t)n * DIM + d];
    __bf16 h = (__bf16)v;
    Bh[(size_t)n * DIM + d] = h;
    Bl[(size_t)n * DIM + d] = (__bf16)(v - (float)h);
    ssum += v * v;
  }
  red[t] = ssum;
  __syncthreads();
  for (int s = 128; s > 0; s >>= 1) {
    if (t < s) red[t] += red[t + s];
    __syncthreads();
  }
  if (t == 0) csq[n] = red[0];
}

// ----------------- TDM descriptor build + issue (ISA §8.3/8.4) -------------
// 2D tile load: tile_dim0=BK elems along contiguous K, tile_dim1=rows,
// tensor_dim0_stride=DIM, data_size=2B, LDS pad: 4 DW after every 16 DW.
// Toolchain here is the clang-23 lane: 6-arg builtin
//   (u32x4 g0, i32x8 g1, i32x4 g2, i32x4 g3, i32x8 g4, i32 cpol).

__device__ __forceinline__ void tdm_load_2d(unsigned lds_off,
                                            const __bf16* gbase,
                                            unsigned tensor_rows,
                                            unsigned tile_rows) {
  unsigned long long ga = (unsigned long long)(uintptr_t)gbase;
  u32x4 g0;
  g0[0] = 1u;                                    // count=1 (valid user D#)
  g0[1] = lds_off;                               // LDS byte address
  g0[2] = (unsigned)(ga & 0xffffffffu);          // global_addr[31:0]
  g0[3] = (unsigned)((ga >> 32) & 0x1ffffffu)    // global_addr[56:32]
          | (2u << 30);                          // type=2 ("image")
  i32x8 g1;
  // data_size=1 (2B) | pad_enable | pad_interval=3 (16 DW) | pad_amount=3 (4 DW)
  g1[0] = (1 << 16) | (1 << 20) | (3 << 22) | (3 << 25);
  const unsigned td0 = DIM;                      // tensor_dim0
  const unsigned td1 = tensor_rows;              // tensor_dim1
  g1[1] = (int)((td0 & 0xffffu) << 16);                          // td0[15:0]
  g1[2] = (int)(((td0 >> 16) & 0xffffu) | ((td1 & 0xffffu) << 16));
  g1[3] = (int)(((td1 >> 16) & 0xffffu) | ((unsigned)BK << 16)); // tile_dim0
  g1[4] = (int)(tile_rows & 0xffffu);            // tile_dim1 (tile_dim2=0)
  g1[5] = (int)DIM;                              // tensor_dim0_stride[31:0]
  g1[6] = 0;
  g1[7] = 0;
  const i32x4 gz  = {0, 0, 0, 0};                // 2D: groups 2/3 unused
  const i32x8 gz8 = {0, 0, 0, 0, 0, 0, 0, 0};    // extra group (unused)
  __builtin_amdgcn_tensor_load_to_lds(g0, g1, gz, gz, gz8, 0);
}

// --------------------------- main fused kernel -----------------------------

__global__ __launch_bounds__(256) void vq_wmma_kernel(
    const __bf16* __restrict__ Ah, const __bf16* __restrict__ Al,
    const __bf16* __restrict__ Bh, const __bf16* __restrict__ Bl,
    const float* __restrict__ csq, const float* __restrict__ emb,
    const float* __restrict__ pe, float* __restrict__ out) {

  // Raw LDS region: GEMM double buffers overlap the post-loop argmin scratch.
  __shared__ alignas(16) unsigned char smem[SMEM_BYTES];

  const int tid  = threadIdx.x;
  const int lane = tid & 31;
  const int wave = tid >> 5;
  const int wm   = wave & 1;   // 2 wave-rows  (32 rows each)
  const int wn   = wave >> 1;  // 4 wave-cols  (32 cols each)
  const int l15  = lane & 15;
  const int lhi  = lane >> 4;  // lane half (ISA 7.12.2 wave32 layouts)
  const int row0 = blockIdx.x * BM;

  unsigned char* const asBase = smem;
  unsigned char* const bsBase = smem + 4 * ASIZE_PLANE;
#define APLANE(p, v) (asBase + (size_t)(((p) * 2 + (v)) * ASIZE_PLANE))
#define BPLANE(p, v) (bsBase + (size_t)(((p) * 2 + (v)) * BSIZE_PLANE))

  v8f acc[2][2];
  // Per-lane running argmin. C-fragment layout: VGPR slot i of lane l maps to
  // a fixed row (i + 8*lhi) over the N loop; column varies -> register trackers.
  float minv[16];
  int   mini[16];
#pragma unroll
  for (int t2 = 0; t2 < 16; ++t2) { minv[t2] = __builtin_inff(); mini[t2] = 0; }

  const v8f vzero = {0.f, 0.f, 0.f, 0.f, 0.f, 0.f, 0.f, 0.f};

  for (int nt = 0; nt < NCODES / BN; ++nt) {
#pragma unroll
    for (int mi = 0; mi < 2; ++mi)
#pragma unroll
      for (int ni = 0; ni < 2; ++ni) acc[mi][ni] = vzero;

    const int ncol0 = nt * BN;

    // ---- prologue: TDM-load buffer 0 (chunk ks=0) ----
    if (wave == 0) {
      tdm_load_2d((unsigned)(uintptr_t)APLANE(0, 0),
                  Ah + (size_t)row0 * DIM, M_TOTAL, BM);
      tdm_load_2d((unsigned)(uintptr_t)APLANE(0, 1),
                  Al + (size_t)row0 * DIM, M_TOTAL, BM);
      tdm_load_2d((unsigned)(uintptr_t)BPLANE(0, 0),
                  Bh + (size_t)ncol0 * DIM, NCODES, BN);
      tdm_load_2d((unsigned)(uintptr_t)BPLANE(0, 1),
                  Bl + (size_t)ncol0 * DIM, NCODES, BN);
    }
    __builtin_amdgcn_s_wait_tensorcnt(0);
    __syncthreads();

    for (int ks = 0; ks < DIM / BK; ++ks) {
      const int p = ks & 1;

      // ---- TDM-prefetch next K chunk into the other buffer ----
      if (wave == 0 && ks + 1 < DIM / BK) {
        const int kb1 = (ks + 1) * BK;
        tdm_load_2d((unsigned)(uintptr_t)APLANE(p ^ 1, 0),
                    Ah + (size_t)row0 * DIM + kb1, M_TOTAL, BM);
        tdm_load_2d((unsigned)(uintptr_t)APLANE(p ^ 1, 1),
                    Al + (size_t)row0 * DIM + kb1, M_TOTAL, BM);
        tdm_load_2d((unsigned)(uintptr_t)BPLANE(p ^ 1, 0),
                    Bh + (size_t)ncol0 * DIM + kb1, NCODES, BN);
        tdm_load_2d((unsigned)(uintptr_t)BPLANE(p ^ 1, 1),
                    Bl + (size_t)ncol0 * DIM + kb1, NCODES, BN);
      }

      // ---- load fragments from current buffer (ISA 7.12.2 layouts) ----
      // A 16x32 bf16: lane l: row=l15, K = {lhi*8..+7} and {lhi*8+16..+23}
      Frag afh[2], afl[2], bfh[2], bfl[2];
#pragma unroll
      for (int mi = 0; mi < 2; ++mi) {
        const int row = wm * 32 + mi * 16 + l15;
        const int kh  = lhi * 8;
        const unsigned char* a0 = APLANE(p, 0) + (size_t)(row * A_LD + kh) * 2;
        const unsigned char* a1 = APLANE(p, 1) + (size_t)(row * A_LD + kh) * 2;
        afh[mi].q[0] = *(const uint4*)(a0);
        afh[mi].q[1] = *(const uint4*)(a0 + 32);
        afl[mi].q[0] = *(const uint4*)(a1);
        afl[mi].q[1] = *(const uint4*)(a1 + 32);
      }
      // B 32x16 bf16: lane l: col=l15, K = {lhi*8..+7} and {lhi*8+16..+23}
#pragma unroll
      for (int ni = 0; ni < 2; ++ni) {
        const int col = wn * 32 + ni * 16 + l15;
        const int kc  = lhi * 8;
        const unsigned char* b0 = BPLANE(p, 0) + (size_t)(col * B_LD + kc) * 2;
        const unsigned char* b1 = BPLANE(p, 1) + (size_t)(col * B_LD + kc) * 2;
        bfh[ni].q[0] = *(const uint4*)(b0);
        bfh[ni].q[1] = *(const uint4*)(b0 + 32);
        bfl[ni].q[0] = *(const uint4*)(b1);
        bfl[ni].q[1] = *(const uint4*)(b1 + 32);
      }

      // ---- bf16x3 MMA: 12 v_wmma per K-step per wave ----
#pragma unroll
      for (int mi = 0; mi < 2; ++mi)
#pragma unroll
        for (int ni = 0; ni < 2; ++ni) {
          acc[mi][ni] = __builtin_amdgcn_wmma_f32_16x16x32_bf16(
              false, afh[mi].v, false, bfh[ni].v, (short)0, acc[mi][ni],
              false, false);
          acc[mi][ni] = __builtin_amdgcn_wmma_f32_16x16x32_bf16(
              false, afh[mi].v, false, bfl[ni].v, (short)0, acc[mi][ni],
              false, false);
          acc[mi][ni] = __builtin_amdgcn_wmma_f32_16x16x32_bf16(
              false, afl[mi].v, false, bfh[ni].v, (short)0, acc[mi][ni],
              false, false);
        }

      // next buffer must be resident before the next iteration reads it
      __builtin_amdgcn_s_wait_tensorcnt(0);
      __syncthreads();
    }

    // ---- fused argmin epilogue for this N tile (registers only) ----
#pragma unroll
    for (int ni = 0; ni < 2; ++ni) {
      const int col = ncol0 + wn * 32 + ni * 16 + l15;
      const float cs = csq[col];
#pragma unroll
      for (int mi = 0; mi < 2; ++mi)
#pragma unroll
        for (int i = 0; i < 8; ++i) {
          float sc = fmaf(-2.f, acc[mi][ni][i], cs);
          const int t2 = mi * 8 + i;
          if (sc < minv[t2] || (sc == minv[t2] && col < mini[t2])) {
            minv[t2] = sc;
            mini[t2] = col;
          }
        }
    }
  }

  // ---- cross-lane / cross-wave argmin reduction (reuses GEMM LDS) ----
  float* redv = (float*)smem;                       // [BM][64]
  int*   redi = (int*)(smem + BM * 64 * 4);         // [BM][64]
  int*   best = (int*)(smem + 2 * BM * 64 * 4);     // [BM]
  __syncthreads();
#pragma unroll
  for (int mi = 0; mi < 2; ++mi)
#pragma unroll
    for (int i = 0; i < 8; ++i) {
      const int r = wm * 32 + mi * 16 + i + lhi * 8;  // WG-local row
      const int c = wn * 16 + l15;                    // candidate slot
      redv[r * 64 + c] = minv[mi * 8 + i];
      redi[r * 64 + c] = mini[mi * 8 + i];
    }
  __syncthreads();
  if (tid < BM) {
    float bv = redv[tid * 64];
    int   bi = redi[tid * 64];
    for (int c = 1; c < 64; ++c) {
      const float v  = redv[tid * 64 + c];
      const int   ix = redi[tid * 64 + c];
      if (v < bv || (v == bv && ix < bi)) { bv = v; bi = ix; }
    }
    best[tid] = bi;
  }
  __syncthreads();

  // ---- gather emb[idx] + pe and write output ----
  const float4* emb4 = (const float4*)emb;
  const float4* pe4  = (const float4*)pe;
  float4*       out4 = (float4*)out;
  for (int e = tid; e < BM * (DIM / 4); e += 256) {
    const int r   = e >> 7;        // DIM/4 == 128
    const int d4  = e & 127;
    const int row = row0 + r;
    const int s   = row >> 3;      // row = s*BATCH + b
    const int idx = best[r];
    const float4 ev = emb4[(size_t)idx * 128 + d4];
    const float4 pv = pe4[(size_t)s * 128 + d4];
    float4 o;
    o.x = ev.x + pv.x; o.y = ev.y + pv.y;
    o.z = ev.z + pv.z; o.w = ev.w + pv.w;
    out4[(size_t)row * 128 + d4] = o;
  }
#undef APLANE
#undef BPLANE
}

// --------------------------- launcher --------------------------------------

extern "C" void kernel_launch(void* const* d_in, const int* in_sizes, int n_in,
                              void* d_out, int out_size, void* d_ws,
                              size_t ws_size, hipStream_t stream) {
  const float* x   = (const float*)d_in[0];  // [2048, 8, 512]
  const float* cb  = (const float*)d_in[1];  // [8192, 512]
  const float* emb = (const float*)d_in[2];  // [8192, 512]
  const float* pe  = (const float*)d_in[3];  // [2048, 1, 512]

  // workspace layout (bytes): Ah 16MB | Al 16MB | Bh 8MB | Bl 8MB | csq 32KB
  char* ws = (char*)d_ws;
  const size_t ABYTES = (size_t)M_TOTAL * DIM * 2;  // bf16 plane of x
  const size_t BBYTES = (size_t)NCODES * DIM * 2;   // bf16 plane of codebook
  __bf16* Ah  = (__bf16*)(ws);
  __bf16* Al  = (__bf16*)(ws + ABYTES);
  __bf16* Bh  = (__bf16*)(ws + 2 * ABYTES);
  __bf16* Bl  = (__bf16*)(ws + 2 * ABYTES + BBYTES);
  float*  Csq = (float*)(ws + 2 * ABYTES + 2 * BBYTES);

  split_x_kernel<<<2048, 256, 0, stream>>>(x, Ah, Al, M_TOTAL * DIM);
  split_codebook_kernel<<<NCODES, 256, 0, stream>>>(cb, Bh, Bl, Csq);
  vq_wmma_kernel<<<M_TOTAL / BM, 256, 0, stream>>>(Ah, Al, Bh, Bl, Csq, emb,
                                                   pe, (float*)d_out);
}
